// xLSTM_38611755991497
// MI455X (gfx1250) — matrix operand order repaired and checked by reading, not compiled, not run on hardware
//
#include <hip/hip_runtime.h>
#include <math.h>

// ---------------- model constants ----------------
constexpr int Bc    = 8;
constexpr int Sc    = 1024;
constexpr int Fc    = 6;
constexpr int NACT  = 32;
constexpr int DEMB  = 256;
constexpr int Dc    = 512;        // 2*DEMB
constexpr int INNER = 1024;       // 2*D
constexpr int NHc   = 4;
constexpr int DHM   = 256;        // INNER/NH
constexpr int NBc   = 256;        // INNER/4
constexpr int DHS   = 128;        // D/NH
constexpr int UPc   = 704;
constexpr int KC    = 4;          // conv taps
constexpr int BS    = Bc * Sc;    // 8192

// ---------------- types for WMMA ----------------
typedef __attribute__((ext_vector_type(16))) __bf16 v16bf;
typedef __attribute__((ext_vector_type(8)))  float  v8f;
typedef __attribute__((ext_vector_type(4)))  unsigned int u32x4;

union FragU { u32x4 q[2]; v16bf v; };

// ---------------- device helpers ----------------
__device__ __forceinline__ unsigned short f2bf(float f) {
  unsigned int u = __builtin_bit_cast(unsigned int, f);
  unsigned int r = u + 0x7FFFu + ((u >> 16) & 1u);
  return (unsigned short)(r >> 16);
}
__device__ __forceinline__ float sigm(float x) { return 1.0f / (1.0f + expf(-x)); }
__device__ __forceinline__ float swishf(float x) { return x * sigm(x); }
__device__ __forceinline__ float logsig(float x) {
  return fminf(x, 0.0f) - log1pf(expf(-fabsf(x)));
}
__device__ __forceinline__ float gelu_exact(float x) {
  return 0.5f * x * (1.0f + erff(x * 0.7071067811865475f));
}
__device__ __forceinline__ float blk_sum(float v, float* sh, int n) {
  int tid = threadIdx.x;
  sh[tid] = v; __syncthreads();
  for (int o = n >> 1; o > 0; o >>= 1) { if (tid < o) sh[tid] += sh[tid + o]; __syncthreads(); }
  float r = sh[0]; __syncthreads();
  return r;
}
__device__ __forceinline__ float blk_max(float v, float* sh, int n) {
  int tid = threadIdx.x;
  sh[tid] = v; __syncthreads();
  for (int o = n >> 1; o > 0; o >>= 1) { if (tid < o) sh[tid] = fmaxf(sh[tid], sh[tid + o]); __syncthreads(); }
  float r = sh[0]; __syncthreads();
  return r;
}

// ---------------- batched WMMA bf16 GEMM, 2x2 register blocking ----------------
// C[M,N] = alpha * A(M x K, row-major bf16, lda) * Bt(N x K, row-major bf16, ldb)^T
// one 32x32 output block per wave32 (2x2 tiles of 16x16);
// batch base = (batch/ic)*Out + (batch%ic)*In for A/B/C. M,N multiples of 32, K of 32.
__global__ void wmma_gemm(const unsigned short* __restrict__ A,
                          const unsigned short* __restrict__ Bt,
                          float* __restrict__ C,
                          int Kd, int lda, int ldb, int ldc,
                          long long aO, long long aI,
                          long long bO, long long bI,
                          long long cO, long long cI,
                          int ic, float alpha) {
  int batch = blockIdx.z;
  const unsigned short* Ab  = A  + (size_t)(batch / ic) * aO + (size_t)(batch % ic) * aI;
  const unsigned short* Bb  = Bt + (size_t)(batch / ic) * bO + (size_t)(batch % ic) * bI;
  float*                Cb  = C  + (size_t)(batch / ic) * cO + (size_t)(batch % ic) * cI;

  int lane = threadIdx.x;               // 0..31
  int half = lane >> 4;                 // 0 or 1
  int row0 = blockIdx.y * 32 + (lane & 15);   // A rows: row0, row0+16
  int col0 = blockIdx.x * 32 + (lane & 15);   // Bt rows (= C cols): col0, col0+16
  int aoff = half * 8;                  // A: lanes>=16 start at K+8
  int boff = half * 8;                  // B: lanes>=16 start at K+8

  v8f acc00 = {0.f,0.f,0.f,0.f,0.f,0.f,0.f,0.f};
  v8f acc01 = acc00, acc10 = acc00, acc11 = acc00;

  const unsigned short* ap0 = Ab + (size_t)row0 * lda + aoff;
  const unsigned short* ap1 = ap0 + (size_t)16 * lda;
  const unsigned short* bp0 = Bb + (size_t)col0 * ldb + boff;
  const unsigned short* bp1 = bp0 + (size_t)16 * ldb;

  for (int k0 = 0; k0 < Kd; k0 += 32) {
    FragU fa0, fa1, fb0, fb1;
    fa0.q[0] = *(const u32x4*)(const void*)(ap0 + k0);
    fa0.q[1] = *(const u32x4*)(const void*)(ap0 + k0 + 16);   // elems 8..15 = K+16..K+23
    fa1.q[0] = *(const u32x4*)(const void*)(ap1 + k0);
    fa1.q[1] = *(const u32x4*)(const void*)(ap1 + k0 + 16);
    fb0.q[0] = *(const u32x4*)(const void*)(bp0 + k0);
    fb0.q[1] = *(const u32x4*)(const void*)(bp0 + k0 + 16);   // elems 8..15 = K+16..K+23
    fb1.q[0] = *(const u32x4*)(const void*)(bp1 + k0);
    fb1.q[1] = *(const u32x4*)(const void*)(bp1 + k0 + 16);
    acc00 = __builtin_amdgcn_wmma_f32_16x16x32_bf16(false, fa0.v, false, fb0.v, (short)0, acc00, false, false);
    acc01 = __builtin_amdgcn_wmma_f32_16x16x32_bf16(false, fa0.v, false, fb1.v, (short)0, acc01, false, false);
    acc10 = __builtin_amdgcn_wmma_f32_16x16x32_bf16(false, fa1.v, false, fb0.v, (short)0, acc10, false, false);
    acc11 = __builtin_amdgcn_wmma_f32_16x16x32_bf16(false, fa1.v, false, fb1.v, (short)0, acc11, false, false);
  }

  int crow = blockIdx.y * 32 + half * 8;
  {
    float* cp = Cb + (size_t)crow * ldc + col0;
#pragma unroll
    for (int r = 0; r < 8; r++) { cp[(size_t)r * ldc] = acc00[r] * alpha; cp[(size_t)r * ldc + 16] = acc01[r] * alpha; }
  }
  {
    float* cp = Cb + (size_t)(crow + 16) * ldc + col0;
#pragma unroll
    for (int r = 0; r < 8; r++) { cp[(size_t)r * ldc] = acc10[r] * alpha; cp[(size_t)r * ldc + 16] = acc11[r] * alpha; }
  }
}

// ---------------- elementwise / small kernels ----------------
__global__ void embed_kernel(const int* __restrict__ act, const float* __restrict__ tm,
                             const float* __restrict__ embW, const float* __restrict__ timeW,
                             const float* __restrict__ timeB, float* __restrict__ x) {
  int idx = blockIdx.x * blockDim.x + threadIdx.x;
  if (idx >= BS * Dc) return;
  int row = idx >> 9, col = idx & 511;
  if (col < DEMB) {
    x[idx] = embW[(size_t)act[row] * DEMB + col];
  } else {
    int j = col - DEMB;
    float a = timeB[j];
    const float* tp = tm + (size_t)row * Fc;
#pragma unroll
    for (int f = 0; f < Fc; f++) a += tp[f] * timeW[f * DEMB + j];
    x[idx] = a;
  }
}

__global__ void ln_kernel(const float* __restrict__ in, const float* __restrict__ g,
                          float* __restrict__ out, int N) {
  __shared__ float sh[256];
  int row = blockIdx.x, tid = threadIdx.x;
  const float* ip = in + (size_t)row * N;
  float* op = out + (size_t)row * N;
  float s = 0.f;
  for (int i = tid; i < N; i += 256) s += ip[i];
  float mu = blk_sum(s, sh, 256) / N;
  float v = 0.f;
  for (int i = tid; i < N; i += 256) { float d = ip[i] - mu; v += d * d; }
  float rstd = rsqrtf(blk_sum(v, sh, 256) / N + 1e-5f);
  for (int i = tid; i < N; i += 256) op[i] = (ip[i] - mu) * rstd * g[i];
}

__global__ void cvt_kernel(const float* __restrict__ in, unsigned short* __restrict__ out, int n) {
  int i = blockIdx.x * blockDim.x + threadIdx.x;
  if (i < n) out[i] = f2bf(in[i]);
}

// out[c*R + r] = bf16(in[r*C + c])   (weights [K][N] -> [N][K])
__global__ void tcvt_kernel(const float* __restrict__ in, unsigned short* __restrict__ out,
                            int R, int C) {
  int i = blockIdx.x * blockDim.x + threadIdx.x;
  if (i >= R * C) return;
  int r = i / C, c = i % C;
  out[(size_t)c * R + r] = f2bf(in[i]);
}

// causal conv (K=4) + swish;  in row stride ldi, C channels
__global__ void conv_kernel(const float* __restrict__ in, const float* __restrict__ w,
                            const float* __restrict__ bias, float* __restrict__ out,
                            int C, int ldi) {
  int idx = blockIdx.x * blockDim.x + threadIdx.x;
  if (idx >= BS * C) return;
  int row = idx / C, ch = idx % C;
  int b = row / Sc, s = row % Sc;
  float acc = bias[ch];
#pragma unroll
  for (int k = 0; k < KC; k++) {
    int sp = s - (KC - 1) + k;
    if (sp >= 0) acc += in[((size_t)(b * Sc + sp)) * ldi + ch] * w[ch * KC + k];
  }
  out[(size_t)row * C + ch] = swishf(acc);
}

// headwise 4x4 q/k/v
__global__ void qkv_kernel(const float* __restrict__ xc, const float* __restrict__ xi,
                           const float* __restrict__ wq, const float* __restrict__ wk,
                           const float* __restrict__ wv,
                           float* __restrict__ qf, float* __restrict__ kf, float* __restrict__ vf,
                           unsigned short* __restrict__ qbf, unsigned short* __restrict__ kbf) {
  int idx = blockIdx.x * blockDim.x + threadIdx.x;
  if (idx >= BS * NBc) return;
  int bs = idx / NBc, nb = idx % NBc;
  const float* xc4 = xc + (size_t)bs * INNER + nb * 4;
  const float* xm4 = xi + (size_t)bs * (2 * INNER) + nb * 4;   // x_m slice of xi
  size_t obase = (size_t)bs * INNER + nb * 4;
#pragma unroll
  for (int o = 0; o < 4; o++) {
    float aq = 0.f, ak = 0.f, av = 0.f;
#pragma unroll
    for (int d = 0; d < 4; d++) {
      aq += xc4[d] * wq[nb * 16 + o * 4 + d];
      ak += xc4[d] * wk[nb * 16 + o * 4 + d];
      av += xm4[d] * wv[nb * 16 + o * 4 + d];
    }
    qf[obase + o] = aq;  qbf[obase + o] = f2bf(aq);
    kf[obase + o] = ak;  kbf[obase + o] = f2bf(ak);
    vf[obase + o] = av;
  }
}

// ig/fg: (B,S) rows, dot over 3072 concat(q,k,v); outputs transposed (B,NH,S)
__global__ void igfg_kernel(const float* __restrict__ qf, const float* __restrict__ kf,
                            const float* __restrict__ vf,
                            const float* __restrict__ Wig, const float* __restrict__ big,
                            const float* __restrict__ Wfg, const float* __restrict__ bfg,
                            float* __restrict__ igt, float* __restrict__ fgt) {
  __shared__ float sh[256];
  int bs = blockIdx.x, tid = threadIdx.x;
  int b = bs >> 10, s = bs & 1023;
  const float* q = qf + (size_t)bs * INNER;
  const float* k = kf + (size_t)bs * INNER;
  const float* v = vf + (size_t)bs * INNER;
  float aI[4] = {0,0,0,0}, aF[4] = {0,0,0,0};
  for (int i = tid; i < 3 * INNER; i += 256) {
    float x = (i < INNER) ? q[i] : (i < 2 * INNER) ? k[i - INNER] : v[i - 2 * INNER];
    const float* wi = Wig + (size_t)i * NHc;
    const float* wf = Wfg + (size_t)i * NHc;
#pragma unroll
    for (int j = 0; j < 4; j++) { aI[j] += x * wi[j]; aF[j] += x * wf[j]; }
  }
#pragma unroll
  for (int j = 0; j < 4; j++) {
    float si = blk_sum(aI[j], sh, 256);
    if (tid == 0) igt[((size_t)(b * NHc + j)) * Sc + s] = si + big[j];
    float sf = blk_sum(aF[j], sh, 256);
    if (tid == 0) fgt[((size_t)(b * NHc + j)) * Sc + s] = sf + bfg[j];
  }
}

// per-(b,h) cumsum of logsigmoid(fg) over S
__global__ void cum_kernel(const float* __restrict__ fgt, float* __restrict__ cvec) {
  int bh = threadIdx.x;
  if (bh >= Bc * NHc) return;
  float run = 0.f;
  for (int t = 0; t < Sc; t++) {
    run += logsig(fgt[(size_t)bh * Sc + t]);
    cvec[(size_t)bh * Sc + t] = run;
  }
}

// vt[(bh*DHM + d)*S + t] = bf16(v[(b*S+t)*INNER + h*DHM + d])
__global__ void vtrans_kernel(const float* __restrict__ vf, unsigned short* __restrict__ vt) {
  int idx = blockIdx.x * blockDim.x + threadIdx.x;
  if (idx >= Bc * NHc * DHM * Sc) return;
  int t = idx & 1023;
  int r = idx >> 10;
  int d = r & 255; r >>= 8;
  int h = r & 3;   int b = r >> 2;
  vt[idx] = f2bf(vf[((size_t)(b * Sc + t)) * INNER + h * DHM + d]);
}

// mLSTM decay/normalizer row pass; in-place on scores (already qk/sqrt(DHM)); writes bf16 P
__global__ void mlstm_p_kernel(float* __restrict__ scores, const float* __restrict__ cvec,
                               const float* __restrict__ igt, unsigned short* __restrict__ P) {
  __shared__ float sh[256];
  int rid = blockIdx.x;          // (bh<<10)|s
  int s = rid & 1023, bh = rid >> 10;
  int tid = threadIdx.x;
  float* srow = scores + (size_t)rid * Sc;
  const float* cv = cvec + (size_t)bh * Sc;
  const float* iv = igt + (size_t)bh * Sc;
  float cs = cv[s];
  float mloc = -INFINITY;
  for (int t = tid; t <= s; t += 256) mloc = fmaxf(mloc, cs - cv[t] + iv[t]);
  float m = blk_max(mloc, sh, 256);
  float sloc = 0.f;
  for (int t = tid; t <= s; t += 256) {
    float cvl = srow[t] * expf(cs - cv[t] + iv[t] - m);
    srow[t] = cvl;
    sloc += cvl;
  }
  float rs = blk_sum(sloc, sh, 256);
  float norm = fmaxf(fabsf(rs), expf(-m)) + 1e-6f;
  unsigned short* prow = P + (size_t)rid * Sc;
  for (int t = tid; t < Sc; t += 256)
    prow[t] = (t <= s) ? f2bf(srow[t] / norm) : (unsigned short)0;
}

// mh_norm(htil) + skip/gate combine -> hs (bf16 only)
__global__ void hs_kernel(const float* __restrict__ htil, const float* __restrict__ xc,
                          const float* __restrict__ xi, const float* __restrict__ gnorm,
                          const float* __restrict__ skip, unsigned short* __restrict__ hsbf) {
  __shared__ float sh[256];
  int rid = blockIdx.x;          // (bh<<10)|s
  int s = rid & 1023, bh = rid >> 10;
  int h = bh & 3, b = bh >> 2;
  int d = threadIdx.x;           // 0..255
  const float* hp = htil + (size_t)rid * DHM;
  float v = hp[d];
  float mu = blk_sum(v, sh, 256) / DHM;
  float dv = v - mu;
  float var = blk_sum(dv * dv, sh, 256) / DHM;
  float hn = dv * rsqrtf(var + 1e-5f) * gnorm[h * DHM + d];
  int col = h * DHM + d;
  size_t r2 = (size_t)(b * Sc + s);
  float xcv = xc[r2 * INNER + col];
  float zv  = xi[r2 * (2 * INNER) + INNER + col];
  float hsv = (hn + skip[col] * xcv) * swishf(zv);
  hsbf[r2 * INNER + col] = f2bf(hsv);
}

__global__ void add_kernel(float* __restrict__ a, const float* __restrict__ b, int n) {
  int i = blockIdx.x * blockDim.x + threadIdx.x;
  if (i < n) a[i] += b[i];
}

// sLSTM scan: one block per (b,h), 128 threads (one per dhs element)
__global__ void slstm_scan_kernel(const float* __restrict__ gx, const float* __restrict__ R,
                                  const float* __restrict__ bias, float* __restrict__ y_out) {
  __shared__ float ysh[DHS];
  int bh = blockIdx.x;
  int b = bh / NHc, h = bh % NHc;
  int d = threadIdx.x;
  const float* Rh = R + (size_t)h * DHS * 4 * DHS;   // [dp][4*DHS]
  float bv[4];
#pragma unroll
  for (int g = 0; g < 4; g++) bv[g] = bias[(h * 4 + g) * DHS + d];
  float c = 0.f, n = 0.f, m = 0.f, y = 0.f;
  ysh[d] = 0.f;
  __syncthreads();
  for (int s = 0; s < Sc; s++) {
    float rg0 = 0.f, rg1 = 0.f, rg2 = 0.f, rg3 = 0.f;
    for (int dp = 0; dp < DHS; dp++) {
      float yv = ysh[dp];
      const float* Rr = Rh + (size_t)dp * (4 * DHS);
      rg0 += yv * Rr[0 * DHS + d];
      rg1 += yv * Rr[1 * DHS + d];
      rg2 += yv * Rr[2 * DHS + d];
      rg3 += yv * Rr[3 * DHS + d];
    }
    const float* g0 = gx + ((size_t)(b * Sc + s) * 4) * Dc + h * DHS + d;
    float ir = g0[0 * Dc] + rg0 + bv[0];
    float fr = g0[1 * Dc] + rg1 + bv[1];
    float zr = g0[2 * Dc] + rg2 + bv[2];
    float og = g0[3 * Dc] + rg3 + bv[3];
    float lfm  = m + logsig(fr);
    float mnew = fmaxf(ir, lfm);
    float ig   = expf(ir - mnew);
    float fg   = expf(lfm - mnew);
    c = fg * c + ig * tanhf(zr);
    n = fg * n + ig;
    m = mnew;
    y = sigm(og) * c / n;
    y_out[((size_t)(b * Sc + s)) * Dc + h * DHS + d] = y;
    __syncthreads();
    ysh[d] = y;
    __syncthreads();
  }
}

// x += mh_norm(y, s_gn_g); one block per (bs,h), 128 threads
__global__ void ygn_kernel(const float* __restrict__ y, const float* __restrict__ g,
                           float* __restrict__ x) {
  __shared__ float sh[128];
  int rid = blockIdx.x;
  int h = rid & 3, row = rid >> 2;
  int d = threadIdx.x;
  const float* yp = y + (size_t)row * Dc + h * DHS;
  float v = yp[d];
  float mu = blk_sum(v, sh, 128) / DHS;
  float dv = v - mu;
  float var = blk_sum(dv * dv, sh, 128) / DHS;
  x[(size_t)row * Dc + h * DHS + d] += dv * rsqrtf(var + 1e-5f) * g[h * DHS + d];
}

__global__ void geluup_kernel(const float* __restrict__ u, unsigned short* __restrict__ gub) {
  int idx = blockIdx.x * blockDim.x + threadIdx.x;
  if (idx >= BS * UPc) return;
  int row = idx / UPc, j = idx % UPc;
  float g  = u[(size_t)row * (2 * UPc) + j];
  float up = u[(size_t)row * (2 * UPc) + UPc + j];
  gub[idx] = f2bf(gelu_exact(g) * up);
}

__global__ void biasout_kernel(float* __restrict__ out, const float* __restrict__ bias) {
  int idx = blockIdx.x * blockDim.x + threadIdx.x;
  if (idx >= BS * NACT) return;
  out[idx] += bias[idx & (NACT - 1)];
}

// ---------------- host launcher ----------------
extern "C" void kernel_launch(void* const* d_in, const int* in_sizes, int n_in,
                              void* d_out, int out_size, void* d_ws, size_t ws_size,
                              hipStream_t stream) {
  (void)in_sizes; (void)n_in; (void)out_size; (void)ws_size;
  // inputs (setup_inputs dict order)
  const int*   act      = (const int*)  d_in[0];
  const float* tm       = (const float*)d_in[1];
  const float* embW     = (const float*)d_in[2];
  const float* timeW    = (const float*)d_in[3];
  const float* timeB    = (const float*)d_in[4];
  const float* ln0_g    = (const float*)d_in[5];
  const float* m_up_W   = (const float*)d_in[6];
  const float* m_conv_W = (const float*)d_in[7];
  const float* m_conv_b = (const float*)d_in[8];
  const float* m_q_W    = (const float*)d_in[9];
  const float* m_k_W    = (const float*)d_in[10];
  const float* m_v_W    = (const float*)d_in[11];
  const float* m_ig_W   = (const float*)d_in[12];
  const float* m_ig_b   = (const float*)d_in[13];
  const float* m_fg_W   = (const float*)d_in[14];
  const float* m_fg_b   = (const float*)d_in[15];
  const float* m_skip   = (const float*)d_in[16];
  const float* m_norm_g = (const float*)d_in[17];
  const float* m_down_W = (const float*)d_in[18];
  const float* ln1_g    = (const float*)d_in[19];
  const float* s_conv_W = (const float*)d_in[20];
  const float* s_conv_b = (const float*)d_in[21];
  const float* s_i_W    = (const float*)d_in[22];
  const float* s_f_W    = (const float*)d_in[23];
  const float* s_z_W    = (const float*)d_in[24];
  const float* s_o_W    = (const float*)d_in[25];
  const float* s_R      = (const float*)d_in[26];
  const float* s_b      = (const float*)d_in[27];
  const float* s_gn_g   = (const float*)d_in[28];
  const float* ln1f_g   = (const float*)d_in[29];
  const float* ff_up_W  = (const float*)d_in[30];
  const float* ff_down_W= (const float*)d_in[31];
  const float* post_ln_g= (const float*)d_in[32];
  const float* cls_W    = (const float*)d_in[33];
  const float* cls_b    = (const float*)d_in[34];
  float* out = (float*)d_out;

  // workspace carve
  char* base = (char*)d_ws;
  size_t off = 0;
  auto carve = [&](size_t bytes) -> void* {
    void* p = base + off;
    off = (off + bytes + 255) & ~(size_t)255;
    return p;
  };
  float* x     = (float*)carve((size_t)BS * Dc * 4);
  float* hbuf  = (float*)carve((size_t)BS * Dc * 4);
  unsigned short* hbf  = (unsigned short*)carve((size_t)BS * Dc * 2);
  float* xi    = (float*)carve((size_t)BS * 2 * INNER * 4);
  float* xc    = (float*)carve((size_t)BS * INNER * 4);
  float* qf    = (float*)carve((size_t)BS * INNER * 4);
  float* kf    = (float*)carve((size_t)BS * INNER * 4);
  float* vf    = (float*)carve((size_t)BS * INNER * 4);
  unsigned short* qbf  = (unsigned short*)carve((size_t)BS * INNER * 2);
  unsigned short* kbf  = (unsigned short*)carve((size_t)BS * INNER * 2);
  unsigned short* vtbf = (unsigned short*)carve((size_t)BS * INNER * 2);
  float* igt   = (float*)carve((size_t)Bc * NHc * Sc * 4);
  float* fgt   = (float*)carve((size_t)Bc * NHc * Sc * 4);
  float* cvec  = (float*)carve((size_t)Bc * NHc * Sc * 4);
  float* scores= (float*)carve((size_t)Bc * NHc * Sc * Sc * 4);   // 128MB, reused
  unsigned short* Pbf = (unsigned short*)carve((size_t)Bc * NHc * Sc * Sc * 2); // 64MB, reused
  float* htil  = (float*)carve((size_t)BS * INNER * 4);
  unsigned short* hsbf = (unsigned short*)carve((size_t)BS * INNER * 2);
  float* hcf   = (float*)carve((size_t)BS * Dc * 4);
  unsigned short* hcbf = (unsigned short*)carve((size_t)BS * Dc * 2);
  float* yb    = (float*)carve((size_t)BS * Dc * 4);
  unsigned short* upWt   = (unsigned short*)carve((size_t)2 * INNER * Dc * 2);
  unsigned short* downWt = (unsigned short*)carve((size_t)Dc * INNER * 2);
  unsigned short* fupWt  = (unsigned short*)carve((size_t)2 * UPc * Dc * 2);
  unsigned short* fdownWt= (unsigned short*)carve((size_t)Dc * UPc * 2);
  unsigned short* clsWt  = (unsigned short*)carve((size_t)NACT * Dc * 2);
  unsigned short* sW[4];
  const float* sWsrc[4] = { s_i_W, s_f_W, s_z_W, s_o_W };
  for (int g = 0; g < 4; g++) sW[g] = (unsigned short*)carve((size_t)NHc * DHS * DHS * 2);
  // region reuse (lifetimes disjoint):
  float* gx  = scores;                         // 64MB, after attention
  float* uf  = scores + (size_t)16777216;      // +64MB, FFN u (BS*1408 f32)
  unsigned short* gubbf = Pbf;                 // BS*704 bf16

  const int TPB = 256;
  auto nblk = [&](size_t n) { return (unsigned)((n + TPB - 1) / TPB); };
  dim3 wblk(32);
  auto gemm = [&](const unsigned short* A, const unsigned short* Bt, float* C,
                  int M, int N, int Kd, int lda, int ldb, int ldc,
                  long long aO, long long aI, long long bO, long long bI,
                  long long cO, long long cI, int ic, int batches, float alpha) {
    dim3 grid((unsigned)(N / 32), (unsigned)(M / 32), (unsigned)batches);
    wmma_gemm<<<grid, wblk, 0, stream>>>(A, Bt, C, Kd, lda, ldb, ldc,
                                         aO, aI, bO, bI, cO, cI, ic, alpha);
  };

  // ---- weight prep (transpose [K][N] -> [N][K] bf16) ----
  tcvt_kernel<<<nblk((size_t)Dc * 2 * INNER), TPB, 0, stream>>>(m_up_W,   upWt,   Dc,   2 * INNER);
  tcvt_kernel<<<nblk((size_t)INNER * Dc),     TPB, 0, stream>>>(m_down_W, downWt, INNER, Dc);
  tcvt_kernel<<<nblk((size_t)Dc * 2 * UPc),   TPB, 0, stream>>>(ff_up_W,  fupWt,  Dc,   2 * UPc);
  tcvt_kernel<<<nblk((size_t)UPc * Dc),       TPB, 0, stream>>>(ff_down_W,fdownWt,UPc,  Dc);
  tcvt_kernel<<<nblk((size_t)Dc * NACT),      TPB, 0, stream>>>(cls_W,    clsWt,  Dc,   NACT);
  for (int g = 0; g < 4; g++)                 // already [h][o][d] == N x K per head
    cvt_kernel<<<nblk((size_t)NHc * DHS * DHS), TPB, 0, stream>>>(sWsrc[g], sW[g], NHc * DHS * DHS);

  // ---- stage 0: embed, ln0, up-proj ----
  embed_kernel<<<nblk((size_t)BS * Dc), TPB, 0, stream>>>(act, tm, embW, timeW, timeB, x);
  ln_kernel<<<BS, TPB, 0, stream>>>(x, ln0_g, hbuf, Dc);
  cvt_kernel<<<nblk((size_t)BS * Dc), TPB, 0, stream>>>(hbuf, hbf, BS * Dc);
  gemm(hbf, upWt, xi, BS, 2 * INNER, Dc, Dc, Dc, 2 * INNER,
       0, 0, 0, 0, 0, 0, 1, 1, 1.0f);

  // ---- mLSTM branch ----
  conv_kernel<<<nblk((size_t)BS * INNER), TPB, 0, stream>>>(xi, m_conv_W, m_conv_b, xc, INNER, 2 * INNER);
  qkv_kernel<<<nblk((size_t)BS * NBc), TPB, 0, stream>>>(xc, xi, m_q_W, m_k_W, m_v_W, qf, kf, vf, qbf, kbf);
  igfg_kernel<<<BS, TPB, 0, stream>>>(qf, kf, vf, m_ig_W, m_ig_b, m_fg_W, m_fg_b, igt, fgt);
  cum_kernel<<<1, 32, 0, stream>>>(fgt, cvec);
  vtrans_kernel<<<nblk((size_t)Bc * NHc * DHM * Sc), TPB, 0, stream>>>(vf, vtbf);
  // qk^T per (b,h): alpha = 1/sqrt(DHM)
  gemm(qbf, kbf, scores, Sc, Sc, DHM, INNER, INNER, Sc,
       (long long)Sc * INNER, DHM, (long long)Sc * INNER, DHM,
       (long long)NHc * Sc * Sc, (long long)Sc * Sc, NHc, Bc * NHc, 1.0f / 16.0f);
  mlstm_p_kernel<<<Bc * NHc * Sc, TPB, 0, stream>>>(scores, cvec, igt, Pbf);
  // P @ V per (b,h)
  gemm(Pbf, vtbf, htil, Sc, DHM, Sc, Sc, Sc, DHM,
       (long long)Sc * Sc, 0, (long long)DHM * Sc, 0,
       (long long)Sc * DHM, 0, 1, Bc * NHc, 1.0f);
  hs_kernel<<<Bc * NHc * Sc, TPB, 0, stream>>>(htil, xc, xi, m_norm_g, m_skip, hsbf);
  gemm(hsbf, downWt, hbuf, BS, Dc, INNER, INNER, INNER, Dc,
       0, 0, 0, 0, 0, 0, 1, 1, 1.0f);
  add_kernel<<<nblk((size_t)BS * Dc), TPB, 0, stream>>>(x, hbuf, BS * Dc);

  // ---- sLSTM branch ----
  ln_kernel<<<BS, TPB, 0, stream>>>(x, ln1_g, hbuf, Dc);
  cvt_kernel<<<nblk((size_t)BS * Dc), TPB, 0, stream>>>(hbuf, hbf, BS * Dc);
  conv_kernel<<<nblk((size_t)BS * Dc), TPB, 0, stream>>>(hbuf, s_conv_W, s_conv_b, hcf, Dc, Dc);
  cvt_kernel<<<nblk((size_t)BS * Dc), TPB, 0, stream>>>(hcf, hcbf, BS * Dc);
  // gate projections into gx layout [(bs)*4 + gate][h*DHS + d]; i,f from hc; z,o from h
  const unsigned short* gA[4] = { hcbf, hcbf, hbf, hbf };
  for (int g = 0; g < 4; g++)
    gemm(gA[g], sW[g], gx + g * Dc, BS, DHS, DHS, Dc, DHS, 4 * Dc,
         0, DHS, 0, (long long)DHS * DHS, 0, DHS, NHc, NHc, 1.0f);
  slstm_scan_kernel<<<Bc * NHc, DHS, 0, stream>>>(gx, s_R, s_b, yb);
  ygn_kernel<<<BS * NHc, DHS, 0, stream>>>(yb, s_gn_g, x);

  // ---- FFN ----
  ln_kernel<<<BS, TPB, 0, stream>>>(x, ln1f_g, hbuf, Dc);
  cvt_kernel<<<nblk((size_t)BS * Dc), TPB, 0, stream>>>(hbuf, hbf, BS * Dc);
  gemm(hbf, fupWt, uf, BS, 2 * UPc, Dc, Dc, Dc, 2 * UPc,
       0, 0, 0, 0, 0, 0, 1, 1, 1.0f);
  geluup_kernel<<<nblk((size_t)BS * UPc), TPB, 0, stream>>>(uf, gubbf);
  gemm(gubbf, fdownWt, hbuf, BS, Dc, UPc, UPc, UPc, Dc,
       0, 0, 0, 0, 0, 0, 1, 1, 1.0f);
  add_kernel<<<nblk((size_t)BS * Dc), TPB, 0, stream>>>(x, hbuf, BS * Dc);

  // ---- head ----
  ln_kernel<<<BS, TPB, 0, stream>>>(x, post_ln_g, hbuf, Dc);
  cvt_kernel<<<nblk((size_t)BS * Dc), TPB, 0, stream>>>(hbuf, hbf, BS * Dc);
  gemm(hbf, clsWt, out, BS, NACT, Dc, Dc, Dc, NACT,
       0, 0, 0, 0, 0, 0, 1, 1, 1.0f);
  biasout_kernel<<<nblk((size_t)BS * NACT), TPB, 0, stream>>>(out, cls_b);
}
